// GPT_86663850099253
// MI455X (gfx1250) — matrix operand order repaired and checked by reading, not compiled, not run on hardware
//
#include <hip/hip_runtime.h>
#include <math.h>

// GPT-2 small forward for MI455X (gfx1250, wave32, WMMA).
#define LNUM 12
#define CDIM 768
#define HN 12
#define HDIM 64
#define TSEQ 1024
#define BBATCH 2
#define MTOK (BBATCH * TSEQ)   // 2048 tokens
#define NIDIM (4 * CDIM)       // 3072
#define VV 50257

typedef unsigned short u16;
typedef __attribute__((ext_vector_type(16))) __bf16 v16bf;
typedef __attribute__((ext_vector_type(8)))  __bf16 v8bf;
typedef __attribute__((ext_vector_type(8)))  float  v8f;

union FragB {
    v16bf v;
    v8bf  h[2];
    u16   s[16];
};

__device__ __forceinline__ u16 f2bf(float f) {
    union { float f; unsigned u; } cv; cv.f = f;
    unsigned u = cv.u;
    u += 0x7fffu + ((u >> 16) & 1u);   // round-to-nearest-even
    return (u16)(u >> 16);
}

__device__ __forceinline__ v8f wmma_bf16(const FragB& a, const FragB& b, v8f c) {
    // D = A(16x32 bf16) x B(32x16 bf16) + C(16x16 f32)
    return __builtin_amdgcn_wmma_f32_16x16x32_bf16(
        false, a.v, false, b.v, (short)0, c, false, false);
}

// ---------------------------------------------------------------------------
// fp32 -> bf16 transpose: out[n*K + k] = bf16(in[k*N + n])
__global__ void convT_kernel(const float* __restrict__ in, u16* __restrict__ out,
                             int K, int N) {
    long i = (long)blockIdx.x * blockDim.x + threadIdx.x;
    long tot = (long)K * N;
    if (i >= tot) return;
    int k = (int)(i / N), n = (int)(i % N);
    out[(long)n * K + k] = f2bf(in[i]);
}

// ---------------------------------------------------------------------------
// x[token][c] = tok_emb[ids[token]][c] + pos_emb[token % T][c]
__global__ void embed_kernel(const int* __restrict__ ids, const float* __restrict__ tok,
                             const float* __restrict__ pos, float* __restrict__ x) {
    long i = (long)blockIdx.x * blockDim.x + threadIdx.x;
    if (i >= (long)MTOK * CDIM) return;
    int token = (int)(i / CDIM), c = (int)(i % CDIM);
    int t = token % TSEQ;
    x[i] = tok[(long)ids[token] * CDIM + c] + pos[(long)t * CDIM + c];
}

// ---------------------------------------------------------------------------
// LayerNorm over C=768; one wave32 per row, 8 rows per 256-thread block.
// Output is bf16 (feeds WMMA GEMM A operand).
__global__ void ln_kernel(const float* __restrict__ x, const float* __restrict__ w,
                          const float* __restrict__ b, u16* __restrict__ out, float eps) {
    int row  = blockIdx.x * 8 + (threadIdx.x >> 5);
    int lane = threadIdx.x & 31;
    const float* xr = x + (long)row * CDIM;
    float v[24];
    float s = 0.f;
#pragma unroll
    for (int i = 0; i < 24; ++i) { v[i] = xr[lane + i * 32]; s += v[i]; }
#pragma unroll
    for (int o = 16; o > 0; o >>= 1) s += __shfl_xor(s, o, 32);
    float mean = s * (1.0f / CDIM);
    float q = 0.f;
#pragma unroll
    for (int i = 0; i < 24; ++i) { float d = v[i] - mean; q += d * d; }
#pragma unroll
    for (int o = 16; o > 0; o >>= 1) q += __shfl_xor(q, o, 32);
    float rstd = rsqrtf(q * (1.0f / CDIM) + eps);
    u16* orow = out + (long)row * CDIM;
#pragma unroll
    for (int i = 0; i < 24; ++i) {
        int c = lane + i * 32;
        orow[c] = f2bf((v[i] - mean) * rstd * w[c] + b[c]);
    }
}

// ---------------------------------------------------------------------------
// bf16 WMMA GEMM:  D[M,N] = A[M,K] (bf16 row-major) x Bt[N,K]^T (bf16, K contiguous)
// Block: 128 threads = 4 waves, 128x128 tile; each wave 64x64 = 4x4 WMMA accs.
// MODE 0: store bf16         (outH, row-major)
// MODE 1: + bias + residual, store f32 (outF)
// MODE 2: exact GELU, store bf16
// MODE 3: + residual, store f32
// MODE 4: store f32, column-guarded (N not multiple of tile)
// MODE 5: store bf16 TRANSPOSED (outH[col*M + row]) -- for V so attention's
//         P·V B-fragments become contiguous v16bf loads over the key dim.
template <int MODE>
__global__ void __launch_bounds__(128)
gemm_kernel(const u16* __restrict__ A, const u16* __restrict__ Bt,
            float* __restrict__ outF, u16* __restrict__ outH,
            const float* __restrict__ bias, const float* __restrict__ resid,
            int M, int N, int K) {
    const int lane = threadIdx.x & 31;
    const int wid  = threadIdx.x >> 5;
    const int hl   = lane >> 4;       // half-wave select
    const int l16  = lane & 15;
    const int rowBase = blockIdx.y * 128 + (wid >> 1) * 64;
    const int colBase = blockIdx.x * 128 + (wid & 1) * 64;

    v8f acc[4][4];
#pragma unroll
    for (int i = 0; i < 4; ++i)
#pragma unroll
        for (int j = 0; j < 4; ++j)
#pragma unroll
            for (int r = 0; r < 8; ++r) acc[i][j][r] = 0.f;

    for (int k0 = 0; k0 < K; k0 += 32) {
        FragB aF[4], bF[4];
#pragma unroll
        for (int i = 0; i < 4; ++i) {
            // A 16x32 layout: lanes 0-15 rows, K block = hl*16 .. hl*16+15 contiguous
            const u16* p = A + (long)(rowBase + i * 16 + l16) * K + k0 + hl * 16;
            aF[i].h[0] = *reinterpret_cast<const v8bf*>(p);
            aF[i].h[1] = *reinterpret_cast<const v8bf*>(p + 8);
        }
#pragma unroll
        for (int j = 0; j < 4; ++j) {
            int n = colBase + j * 16 + l16;
            if (n < N) {
                // B 32x16 layout: lane = column, K block = hl*16 .. hl*16+15 contiguous
                bF[j].v = *reinterpret_cast<const v16bf*>(Bt + (long)n * K + k0 + hl * 16);
            } else {
#pragma unroll
                for (int e = 0; e < 16; ++e) bF[j].s[e] = 0;
            }
        }
        // prefetch next K tile (lowers to global_prefetch_b8)
        if (k0 + 32 < K) {
            __builtin_prefetch((const void*)(A + (long)(rowBase + l16) * K + k0 + 32), 0, 0);
            int np = colBase + l16;
            if (np < N)
                __builtin_prefetch((const void*)(Bt + (long)np * K + k0 + 32), 0, 0);
        }
#pragma unroll
        for (int i = 0; i < 4; ++i)
#pragma unroll
            for (int j = 0; j < 4; ++j)
                acc[i][j] = wmma_bf16(aF[i], bF[j], acc[i][j]);
    }

    // Epilogue: C/D layout — VGPR r holds row (r + 8*hl), lane 0..15 = col.
#pragma unroll
    for (int i = 0; i < 4; ++i)
#pragma unroll
        for (int j = 0; j < 4; ++j)
#pragma unroll
            for (int r = 0; r < 8; ++r) {
                int row = rowBase + i * 16 + r + hl * 8;
                int col = colBase + j * 16 + l16;
                if (MODE == 4 && col >= N) continue;
                long idx = (long)row * N + col;
                float val = acc[i][j][r];
                if (MODE == 0) {
                    outH[idx] = f2bf(val);
                } else if (MODE == 1) {
                    outF[idx] = val + bias[col] + resid[idx];
                } else if (MODE == 2) {
                    val = 0.5f * val * (1.0f + erff(val * 0.70710678118654752f));
                    outH[idx] = f2bf(val);
                } else if (MODE == 3) {
                    outF[idx] = val + resid[idx];
                } else if (MODE == 5) {
                    outH[(long)col * M + row] = f2bf(val);
                } else {
                    outF[idx] = val;
                }
            }
}

// ---------------------------------------------------------------------------
// Causal flash attention. One wave32 block per (batch, head, 16-row Q tile).
// S = Q·K^T via WMMA (contraction over contiguous head-dim), online softmax in
// C/D register layout, P re-shaped through LDS into A layout, O += P·V via
// WMMA with V pre-transposed (Vt[d][token]) so B-fragments load contiguously.
__global__ void __launch_bounds__(32)
attn_kernel(const u16* __restrict__ q, const u16* __restrict__ k,
            const u16* __restrict__ vt, u16* __restrict__ y) {
    __shared__ __align__(32) u16 pbuf[16 * 32];
    const int lane = threadIdx.x & 31;
    const int hl = lane >> 4, l16 = lane & 15;
    int gid = blockIdx.x;
    int qt = gid & 63;
    int h  = (gid >> 6) % HN;
    int b  = gid / (64 * HN);
    const int q0 = qt * 16;
    const long hoff = (long)h * HDIM;
    const u16* Qb  = q + (long)b * TSEQ * CDIM + hoff;
    const u16* Kb  = k + (long)b * TSEQ * CDIM + hoff;
    const u16* Vtb = vt + hoff * MTOK + (long)b * TSEQ;  // Vt[d][token]

    FragB qf[2];   // Q 16x64 as two 16x32 A-fragments
#pragma unroll
    for (int f = 0; f < 2; ++f) {
        const u16* p = Qb + (long)(q0 + l16) * CDIM + f * 32 + hl * 16;
        qf[f].h[0] = *reinterpret_cast<const v8bf*>(p);
        qf[f].h[1] = *reinterpret_cast<const v8bf*>(p + 8);
    }

    v8f oA[4];
#pragma unroll
    for (int f = 0; f < 4; ++f)
#pragma unroll
        for (int r = 0; r < 8; ++r) oA[f][r] = 0.f;
    float mrow[8], lrow[8];
#pragma unroll
    for (int r = 0; r < 8; ++r) { mrow[r] = -1e30f; lrow[r] = 0.f; }

    const float scale = 0.125f;              // 1/sqrt(64)
    const int nkt = (q0 + 15) / 32 + 1;      // causal: last key <= q0+15

    for (int kt = 0; kt < nkt; ++kt) {
        const int tk = kt * 32;
        v8f sA[2];
#pragma unroll
        for (int g = 0; g < 2; ++g)
#pragma unroll
            for (int r = 0; r < 8; ++r) sA[g][r] = 0.f;

        // S(16x32) = Q(16x64) x K^T(64x32): B frag lane = key column,
        // contraction (head-dim) contiguous in K's rows.
#pragma unroll
        for (int g = 0; g < 2; ++g)
#pragma unroll
            for (int f = 0; f < 2; ++f) {
                FragB kf;
                kf.v = *reinterpret_cast<const v16bf*>(
                    Kb + (long)(tk + g * 16 + l16) * CDIM + f * 32 + hl * 16);
                sA[g] = wmma_bf16(qf[f], kf, sA[g]);
            }

        // online softmax; each row lives in one VGPR across a 16-lane half
#pragma unroll
        for (int r = 0; r < 8; ++r) {
            const int m = r + hl * 8;
            const int qrow = q0 + m;
            float s0 = sA[0][r] * scale;
            float s1 = sA[1][r] * scale;
            if (tk + l16 > qrow)      s0 = -1e30f;
            if (tk + 16 + l16 > qrow) s1 = -1e30f;
            float mx = fmaxf(s0, s1);
#pragma unroll
            for (int o = 1; o < 16; o <<= 1) mx = fmaxf(mx, __shfl_xor(mx, o, 32));
            float mnew  = fmaxf(mrow[r], mx);
            float alpha = __expf(mrow[r] - mnew);
            float p0 = __expf(s0 - mnew);
            float p1 = __expf(s1 - mnew);
            pbuf[m * 32 + l16]      = f2bf(p0);
            pbuf[m * 32 + 16 + l16] = f2bf(p1);
            float ps = p0 + p1;
#pragma unroll
            for (int o = 1; o < 16; o <<= 1) ps += __shfl_xor(ps, o, 32);
            lrow[r] = lrow[r] * alpha + ps;
            mrow[r] = mnew;
#pragma unroll
            for (int f = 0; f < 4; ++f) oA[f][r] *= alpha;
        }
        __syncthreads();   // single-wave WG: LDS ordering fence

        FragB pf;          // P(16x32) as A-fragment from LDS
        {
            const u16* p = &pbuf[l16 * 32 + hl * 16];
            pf.h[0] = *reinterpret_cast<const v8bf*>(p);
            pf.h[1] = *reinterpret_cast<const v8bf*>(p + 8);
        }
        // O(16x64) += P(16x32) x V(32x64): B frag lane = head-dim column,
        // contraction (keys) contiguous in Vt rows -> one v16bf load per lane.
#pragma unroll
        for (int f = 0; f < 4; ++f) {
            FragB vf;
            const int d = f * 16 + l16;
            vf.v = *reinterpret_cast<const v16bf*>(
                Vtb + (long)d * MTOK + tk + hl * 16);
            oA[f] = wmma_bf16(pf, vf, oA[f]);
        }
        __syncthreads();   // protect pbuf before next iteration rewrites it
    }

#pragma unroll
    for (int f = 0; f < 4; ++f)
#pragma unroll
        for (int r = 0; r < 8; ++r) {
            const int m = r + hl * 8;
            y[((long)b * TSEQ + q0 + m) * CDIM + hoff + f * 16 + l16] =
                f2bf(oA[f][r] / lrow[r]);
        }
}

// ---------------------------------------------------------------------------
// Per-token NLL: lse = max + log(sum exp(x - max)); nll = lse - logit[target]
__global__ void nll_kernel(const float* __restrict__ logits, const int* __restrict__ tgt,
                           float* __restrict__ nll) {
    __shared__ float sh[256];
    const int row = blockIdx.x;
    const float* lr = logits + (long)row * VV;
    float lmax = -1e30f;
    for (int j = threadIdx.x; j < VV; j += 256) lmax = fmaxf(lmax, lr[j]);
    sh[threadIdx.x] = lmax; __syncthreads();
    for (int o = 128; o > 0; o >>= 1) {
        if (threadIdx.x < o) sh[threadIdx.x] = fmaxf(sh[threadIdx.x], sh[threadIdx.x + o]);
        __syncthreads();
    }
    float mv = sh[0]; __syncthreads();
    float ls = 0.f;
    for (int j = threadIdx.x; j < VV; j += 256) ls += __expf(lr[j] - mv);
    sh[threadIdx.x] = ls; __syncthreads();
    for (int o = 128; o > 0; o >>= 1) {
        if (threadIdx.x < o) sh[threadIdx.x] += sh[threadIdx.x + o];
        __syncthreads();
    }
    if (threadIdx.x == 0) nll[row] = mv + logf(sh[0]) - lr[tgt[row]];
}

__global__ void loss_kernel(const float* __restrict__ nll, float* __restrict__ loss) {
    __shared__ float sh[256];
    float s = 0.f;
    for (int j = threadIdx.x; j < MTOK; j += 256) s += nll[j];
    sh[threadIdx.x] = s; __syncthreads();
    for (int o = 128; o > 0; o >>= 1) {
        if (threadIdx.x < o) sh[threadIdx.x] += sh[threadIdx.x + o];
        __syncthreads();
    }
    if (threadIdx.x == 0) *loss = sh[0] * (1.0f / MTOK);
}

// ---------------------------------------------------------------------------
extern "C" void kernel_launch(void* const* d_in, const int* in_sizes, int n_in,
                              void* d_out, int out_size, void* d_ws, size_t ws_size,
                              hipStream_t stream) {
    (void)in_sizes; (void)n_in; (void)out_size; (void)ws_size;
    const int*   ids    = (const int*)d_in[0];
    const int*   tgts   = (const int*)d_in[1];
    const float* tok    = (const float*)d_in[2];
    const float* pos    = (const float*)d_in[3];
    const float* Wq     = (const float*)d_in[4];
    const float* Wk     = (const float*)d_in[5];
    const float* Wv     = (const float*)d_in[6];
    const float* Wo     = (const float*)d_in[7];
    const float* bo     = (const float*)d_in[8];
    const float* ln1w   = (const float*)d_in[9];
    const float* ln1b   = (const float*)d_in[10];
    const float* ln2w   = (const float*)d_in[11];
    const float* ln2b   = (const float*)d_in[12];
    const float* W1     = (const float*)d_in[13];
    const float* W2     = (const float*)d_in[14];
    const float* lnfw   = (const float*)d_in[15];
    const float* lnfb   = (const float*)d_in[16];
    const float* Wout   = (const float*)d_in[17];
    float* outF = (float*)d_out;

    char* wsp = (char*)d_ws;
    auto alloc = [&](size_t bytes) -> void* {
        void* p = (void*)wsp;
        wsp += (bytes + 255) & ~(size_t)255;
        return p;
    };
    const size_t CC  = (size_t)CDIM * CDIM;
    const size_t CNI = (size_t)CDIM * NIDIM;
    u16* WqT   = (u16*)alloc(LNUM * CC * 2);
    u16* WkT   = (u16*)alloc(LNUM * CC * 2);
    u16* WvT   = (u16*)alloc(LNUM * CC * 2);
    u16* WoT   = (u16*)alloc(LNUM * CC * 2);
    u16* W1T   = (u16*)alloc(LNUM * CNI * 2);
    u16* W2T   = (u16*)alloc(LNUM * CNI * 2);
    u16* WoutT = (u16*)alloc((size_t)VV * CDIM * 2);
    float* x   = (float*)alloc((size_t)MTOK * CDIM * 4);
    u16* hbf   = (u16*)alloc((size_t)MTOK * CDIM * 2);
    u16* qbf   = (u16*)alloc((size_t)MTOK * CDIM * 2);
    u16* kbf   = (u16*)alloc((size_t)MTOK * CDIM * 2);
    u16* vtbf  = (u16*)alloc((size_t)MTOK * CDIM * 2);   // V transposed [C][MTOK]
    u16* ybf   = (u16*)alloc((size_t)MTOK * CDIM * 2);
    u16* actbf = (u16*)alloc((size_t)MTOK * NIDIM * 2);
    float* nllbuf = (float*)alloc((size_t)MTOK * 4);

    // ---- weight conversion (fp32 -> bf16, transposed) ----
    {
        int gCC  = (int)((CC  + 255) / 256);
        int gCNI = (int)((CNI + 255) / 256);
        for (int l = 0; l < LNUM; ++l) {
            convT_kernel<<<gCC, 256, 0, stream>>>(Wq + l * CC, WqT + l * CC, CDIM, CDIM);
            convT_kernel<<<gCC, 256, 0, stream>>>(Wk + l * CC, WkT + l * CC, CDIM, CDIM);
            convT_kernel<<<gCC, 256, 0, stream>>>(Wv + l * CC, WvT + l * CC, CDIM, CDIM);
            convT_kernel<<<gCC, 256, 0, stream>>>(Wo + l * CC, WoT + l * CC, CDIM, CDIM);
            convT_kernel<<<gCNI, 256, 0, stream>>>(W1 + l * CNI, W1T + l * CNI, CDIM, NIDIM);
            convT_kernel<<<gCNI, 256, 0, stream>>>(W2 + l * CNI, W2T + l * CNI, NIDIM, CDIM);
        }
        long totV = (long)CDIM * VV;
        convT_kernel<<<(int)((totV + 255) / 256), 256, 0, stream>>>(Wout, WoutT, CDIM, VV);
    }

    // ---- embedding ----
    {
        long tot = (long)MTOK * CDIM;
        embed_kernel<<<(int)((tot + 255) / 256), 256, 0, stream>>>(ids, tok, pos, x);
    }

    const dim3 blk128(128);
    const dim3 gC(CDIM / 128, MTOK / 128);      // N=768 tiles
    const dim3 gNI(NIDIM / 128, MTOK / 128);    // N=3072 tiles
    for (int l = 0; l < LNUM; ++l) {
        ln_kernel<<<MTOK / 8, 256, 0, stream>>>(x, ln1w + l * CDIM, ln1b + l * CDIM, hbf, 1e-5f);
        gemm_kernel<0><<<gC, blk128, 0, stream>>>(hbf, WqT + l * CC, nullptr, qbf,
                                                  nullptr, nullptr, MTOK, CDIM, CDIM);
        gemm_kernel<0><<<gC, blk128, 0, stream>>>(hbf, WkT + l * CC, nullptr, kbf,
                                                  nullptr, nullptr, MTOK, CDIM, CDIM);
        gemm_kernel<5><<<gC, blk128, 0, stream>>>(hbf, WvT + l * CC, nullptr, vtbf,
                                                  nullptr, nullptr, MTOK, CDIM, CDIM);
        attn_kernel<<<BBATCH * HN * (TSEQ / 16), 32, 0, stream>>>(qbf, kbf, vtbf, ybf);
        gemm_kernel<1><<<gC, blk128, 0, stream>>>(ybf, WoT + l * CC, x, nullptr,
                                                  bo + l * CDIM, x, MTOK, CDIM, CDIM);
        ln_kernel<<<MTOK / 8, 256, 0, stream>>>(x, ln2w + l * CDIM, ln2b + l * CDIM, hbf, 1e-5f);
        gemm_kernel<2><<<gNI, blk128, 0, stream>>>(hbf, W1T + l * CNI, nullptr, actbf,
                                                   nullptr, nullptr, MTOK, NIDIM, CDIM);
        gemm_kernel<3><<<gC, blk128, 0, stream>>>(actbf, W2T + l * CNI, x, nullptr,
                                                  nullptr, x, MTOK, CDIM, NIDIM);
    }

    // ---- final LN (eps=0 quirk), logits, loss ----
    ln_kernel<<<MTOK / 8, 256, 0, stream>>>(x, lnfw, lnfb, hbf, 0.0f);
    const dim3 gV((VV + 127) / 128, MTOK / 128);
    gemm_kernel<4><<<gV, blk128, 0, stream>>>(hbf, WoutT, outF, nullptr,
                                              nullptr, nullptr, MTOK, VV, CDIM);
    nll_kernel<<<MTOK, 256, 0, stream>>>(outF, tgts, nllbuf);
    loss_kernel<<<1, 256, 0, stream>>>(nllbuf, outF + (size_t)MTOK * VV);
}
